// SS2D_84825604096648
// MI455X (gfx1250) — matrix operand order repaired and checked by reading, not compiled, not run on hardware
//
#include <hip/hip_runtime.h>
#include <hip/hip_bf16.h>
#include <math.h>

typedef __attribute__((ext_vector_type(16))) __bf16 v16bf;
typedef __attribute__((ext_vector_type(8)))  float  v8f;

#define NB   4      // batch
#define HH   32
#define WW   32
#define L0   1024   // H*W
#define DM   512    // d_model
#define DI   1024   // d_inner
#define NSTATE 16
#define DTR  32
#define NK   4

// direction permutation: xs[b,k,d,l] = conv[b, perm_l0(k,l), d]; same map for the
// inverse combine (y[b, perm_l0(k,l), d] += ys[b,k,d,l])
__device__ __forceinline__ int perm_l0(int k, int l) {
  switch (k) {
    case 0: return l;
    case 1: return (l & 31) * 32 + (l >> 5);
    case 2: return 1023 - l;
    default: { int lf = 1023 - l; return (lf & 31) * 32 + (lf >> 5); }
  }
}

// One wave computes a 16x64 f32 strip: 4 accumulators share one A fragment.
// Arow: base of this lane's A row (m fixed, contiguous K)
// Brow[q]: base of this lane's B column source for n-tile q (n fixed, contiguous K)
__device__ __forceinline__ void wmma_row4(const __bf16* __restrict__ Arow,
                                          const __bf16* __restrict__ Brow0,
                                          const __bf16* __restrict__ Brow1,
                                          const __bf16* __restrict__ Brow2,
                                          const __bf16* __restrict__ Brow3,
                                          int Kdim, int laneHalf, v8f acc[4]) {
  for (int k0 = 0; k0 < Kdim; k0 += 32) {
    v16bf a, b0, b1, b2, b3;
#pragma unroll
    for (int i = 0; i < 8; ++i) {             // A 16x32 bf16 layout (contiguous k = 16*half + s)
      a[i]     = Arow[k0 + laneHalf * 16 + i];
      a[i + 8] = Arow[k0 + laneHalf * 16 + 8 + i];
    }
#pragma unroll
    for (int i = 0; i < 16; ++i) {            // B 32x16 bf16 layout
      int kk = k0 + laneHalf * 16 + i;
      b0[i] = Brow0[kk]; b1[i] = Brow1[kk]; b2[i] = Brow2[kk]; b3[i] = Brow3[kk];
    }
    acc[0] = __builtin_amdgcn_wmma_f32_16x16x32_bf16(false, a, false, b0, (short)0, acc[0], false, false);
    acc[1] = __builtin_amdgcn_wmma_f32_16x16x32_bf16(false, a, false, b1, (short)0, acc[1], false, false);
    acc[2] = __builtin_amdgcn_wmma_f32_16x16x32_bf16(false, a, false, b2, (short)0, acc[2], false, false);
    acc[3] = __builtin_amdgcn_wmma_f32_16x16x32_bf16(false, a, false, b3, (short)0, acc[3], false, false);
  }
}

__global__ void cvt_bf16(const float* __restrict__ in, __bf16* __restrict__ out, int n) {
  int i = blockIdx.x * blockDim.x + threadIdx.x;
  if (i < n) out[i] = (__bf16)in[i];
}

__global__ void zero_f32(float* __restrict__ p, int n) {
  int i = blockIdx.x * blockDim.x + threadIdx.x;
  if (i < n) p[i] = 0.f;
}

// xz = x @ in_proj_w^T ; split into x1 (cols<DI) and z. One wave -> 16x64 strip.
__global__ void gemm_inproj(const __bf16* __restrict__ Xbf, const __bf16* __restrict__ Wbf,
                            float* __restrict__ x1, float* __restrict__ z) {
  const int lane = threadIdx.x & 31, wid = threadIdx.x >> 5;
  const int NT4 = (2 * DI) / 64, MT = (NB * L0) / 16;    // 32, 256
  int t = blockIdx.x * 8 + wid;
  if (t >= MT * NT4) return;                             // wave-uniform
  int mt = t / NT4, nt4 = t % NT4;
  int laneHalf = lane >> 4, laneRow = lane & 15;
  int m = mt * 16 + laneRow;
  const __bf16* Arow = Xbf + (size_t)m * DM;
  const __bf16* Br[4];
#pragma unroll
  for (int q = 0; q < 4; ++q)
    Br[q] = Wbf + (size_t)((nt4 * 4 + q) * 16 + laneRow) * DM;
  v8f acc[4] = {};
  wmma_row4(Arow, Br[0], Br[1], Br[2], Br[3], DM, laneHalf, acc);
#pragma unroll
  for (int q = 0; q < 4; ++q) {
    int col = (nt4 * 4 + q) * 16 + laneRow;
    float* dst = (col < DI) ? x1 : z;
    int cj = (col < DI) ? col : col - DI;
#pragma unroll
    for (int r = 0; r < 8; ++r) {
      int row = mt * 16 + laneHalf * 8 + r;
      dst[(size_t)row * DI + cj] = acc[q][r];
    }
  }
}

// depthwise 3x3 + bias + SiLU -> f32 and bf16 copies, layout (b, l0, d)
__global__ void conv_silu(const float* __restrict__ x1, const float* __restrict__ cw,
                          const float* __restrict__ cb,
                          float* __restrict__ cf, __bf16* __restrict__ cbf) {
  int bl = blockIdx.x;                 // b*L0 + l0
  int b = bl >> 10, l = bl & 1023;
  int h = l >> 5, w = l & 31;
  for (int d = threadIdx.x; d < DI; d += blockDim.x) {
    float s = cb[d];
#pragma unroll
    for (int ky = 0; ky < 3; ++ky) {
      int hy = h + ky - 1;
      if (hy < 0 || hy >= HH) continue;
#pragma unroll
      for (int kx = 0; kx < 3; ++kx) {
        int wx = w + kx - 1;
        if (wx < 0 || wx >= WW) continue;
        s += cw[d * 9 + ky * 3 + kx] * x1[((size_t)(b * L0 + hy * 32 + wx)) * DI + d];
      }
    }
    float v = s / (1.f + __expf(-s));
    cf[(size_t)bl * DI + d]  = v;
    cbf[(size_t)bl * DI + d] = (__bf16)v;
  }
}

// x_dbl[b,k,c,l] = sum_d x_proj[k,c,d] * conv[b, perm(k,l), d]; 16x64 strip/wave
__global__ void gemm_xdbl(const __bf16* __restrict__ xproj_bf, const __bf16* __restrict__ cbf,
                          float* __restrict__ xdbl) {
  const int lane = threadIdx.x & 31, wid = threadIdx.x >> 5;
  const int tilesPerBK = 4 * 16;                        // M=64 -> 4, N=1024 -> 16 strips
  int t = blockIdx.x * 8 + wid;
  if (t >= 16 * tilesPerBK) return;                     // wave-uniform
  int bk = t / tilesPerBK, rem = t % tilesPerBK;
  int mt = rem / 16, nt4 = rem % 16;
  int b = bk >> 2, k = bk & 3;
  int laneHalf = lane >> 4, laneRow = lane & 15;
  int m = mt * 16 + laneRow;                            // c index
  const __bf16* Arow = xproj_bf + ((size_t)k * 64 + m) * DI;
  const __bf16* Br[4];
#pragma unroll
  for (int q = 0; q < 4; ++q) {
    int lg = (nt4 * 4 + q) * 16 + laneRow;              // l index
    Br[q] = cbf + ((size_t)(b * L0 + perm_l0(k, lg))) * DI;
  }
  v8f acc[4] = {};
  wmma_row4(Arow, Br[0], Br[1], Br[2], Br[3], DI, laneHalf, acc);
#pragma unroll
  for (int q = 0; q < 4; ++q)
#pragma unroll
    for (int r = 0; r < 8; ++r) {
      int c = mt * 16 + laneHalf * 8 + r;
      xdbl[((size_t)bk * 64 + c) * L0 + (nt4 * 4 + q) * 16 + laneRow] = acc[q][r];
    }
}

// selective scan, one thread per (b,k,d); delta computed on the fly from dts;
// output (incl. D*u skip) combined across the 4 directions via atomicAdd
__global__ void scan_kernel(const float* __restrict__ xdbl, const float* __restrict__ convf,
                            const float* __restrict__ dtw, const float* __restrict__ dtb,
                            const float* __restrict__ Alogs, const float* __restrict__ Dsv,
                            float* __restrict__ ybuf) {
  __shared__ float sm[64 * 16];        // rows: 0..31 dts, 32..47 Bs, 48..63 Cs
  int blk = blockIdx.x;
  int bk = blk >> 2, dblk = blk & 3;
  int b = bk >> 2, k = bk & 3;
  int d = dblk * 256 + threadIdx.x;

  float wreg[DTR];
#pragma unroll
  for (int r = 0; r < DTR; ++r) wreg[r] = dtw[((size_t)(k * DI + d)) * DTR + r];
  float An[NSTATE];
#pragma unroll
  for (int n = 0; n < NSTATE; ++n) An[n] = -__expf(Alogs[d * NSTATE + n]);
  float dtb2 = 2.f * dtb[k * DI + d];  // reference adds dt_b twice
  float Dd = Dsv[k * DI + d];
  float h[NSTATE];
#pragma unroll
  for (int n = 0; n < NSTATE; ++n) h[n] = 0.f;

  const float* xd = xdbl + (size_t)bk * 64 * L0;
  for (int lc = 0; lc < L0; lc += 16) {
    __syncthreads();
    for (int i = threadIdx.x; i < 64 * 16; i += 256) {
      int c = i >> 4, j = i & 15;
      sm[i] = xd[(size_t)c * L0 + lc + j];
    }
    __syncthreads();
    for (int j = 0; j < 16; ++j) {
      int l = lc + j;
      float s = dtb2;
#pragma unroll
      for (int r = 0; r < DTR; ++r) s += wreg[r] * sm[r * 16 + j];
      float delta = (s > 20.f) ? s : log1pf(__expf(s));
      int pl = perm_l0(k, l);
      float u = convf[((size_t)(b * L0 + pl)) * DI + d];
      float du = delta * u;
      float y = 0.f;
#pragma unroll
      for (int n = 0; n < NSTATE; ++n) {
        float dA = __expf(delta * An[n]);
        h[n] = dA * h[n] + du * sm[(32 + n) * 16 + j];
        y += h[n] * sm[(48 + n) * 16 + j];
      }
      y += Dd * u;
      atomicAdd(&ybuf[((size_t)(b * L0 + pl)) * DI + d], y);
    }
  }
}

// LayerNorm over D_INNER, then gate with silu(z), emit bf16
__global__ void ln_gate(const float* __restrict__ ybuf, const float* __restrict__ z,
                        const float* __restrict__ g, const float* __restrict__ bta,
                        __bf16* __restrict__ yg) {
  __shared__ float red[256];
  int row = blockIdx.x;
  float s = 0.f, s2 = 0.f;
  for (int d = threadIdx.x; d < DI; d += 256) {
    float v = ybuf[(size_t)row * DI + d];
    s += v; s2 += v * v;
  }
  red[threadIdx.x] = s; __syncthreads();
  for (int o = 128; o > 0; o >>= 1) { if (threadIdx.x < o) red[threadIdx.x] += red[threadIdx.x + o]; __syncthreads(); }
  float mu = red[0] / DI; __syncthreads();
  red[threadIdx.x] = s2; __syncthreads();
  for (int o = 128; o > 0; o >>= 1) { if (threadIdx.x < o) red[threadIdx.x] += red[threadIdx.x + o]; __syncthreads(); }
  float var = red[0] / DI - mu * mu;
  float inv = rsqrtf(var + 1e-5f);
  for (int d = threadIdx.x; d < DI; d += 256) {
    float v = (ybuf[(size_t)row * DI + d] - mu) * inv * g[d] + bta[d];
    float zz = z[(size_t)row * DI + d];
    float sz = zz / (1.f + __expf(-zz));
    yg[(size_t)row * DI + d] = (__bf16)(v * sz);
  }
}

// out = yg @ out_w^T + out_b ; 16x64 strip per wave
__global__ void gemm_out(const __bf16* __restrict__ yg, const __bf16* __restrict__ Wo,
                         const float* __restrict__ ob, float* __restrict__ out) {
  const int lane = threadIdx.x & 31, wid = threadIdx.x >> 5;
  const int NT4 = DM / 64, MT = (NB * L0) / 16;         // 8, 256
  int t = blockIdx.x * 8 + wid;
  if (t >= MT * NT4) return;                             // wave-uniform
  int mt = t / NT4, nt4 = t % NT4;
  int laneHalf = lane >> 4, laneRow = lane & 15;
  int m = mt * 16 + laneRow;
  const __bf16* Arow = yg + (size_t)m * DI;
  const __bf16* Br[4];
#pragma unroll
  for (int q = 0; q < 4; ++q)
    Br[q] = Wo + (size_t)((nt4 * 4 + q) * 16 + laneRow) * DI;
  v8f acc[4] = {};
  wmma_row4(Arow, Br[0], Br[1], Br[2], Br[3], DI, laneHalf, acc);
#pragma unroll
  for (int q = 0; q < 4; ++q) {
    int col = (nt4 * 4 + q) * 16 + laneRow;
#pragma unroll
    for (int r = 0; r < 8; ++r) {
      int row = mt * 16 + laneHalf * 8 + r;
      out[(size_t)row * DM + col] = acc[q][r] + ob[col];
    }
  }
}

extern "C" void kernel_launch(void* const* d_in, const int* in_sizes, int n_in,
                              void* d_out, int out_size, void* d_ws, size_t ws_size,
                              hipStream_t stream) {
  const float* x_f      = (const float*)d_in[0];
  const float* inw_f    = (const float*)d_in[1];
  const float* convw_f  = (const float*)d_in[2];
  const float* convb_f  = (const float*)d_in[3];
  const float* xproj_f  = (const float*)d_in[4];
  const float* dtw_f    = (const float*)d_in[5];
  const float* dtb_f    = (const float*)d_in[6];
  const float* alogs_f  = (const float*)d_in[7];
  const float* ds_f     = (const float*)d_in[8];
  const float* lng_f    = (const float*)d_in[9];
  const float* lnb_f    = (const float*)d_in[10];
  const float* outw_f   = (const float*)d_in[11];
  const float* outb_f   = (const float*)d_in[12];
  float* out = (float*)d_out;

  char* ws = (char*)d_ws;
  size_t off = 0;
  auto alloc = [&](size_t bytes) { size_t o = off; off = (off + bytes + 255) & ~(size_t)255; return o; };

  __bf16* x_bf     = (__bf16*)(ws + alloc((size_t)NB * L0 * DM * 2));
  __bf16* winbf    = (__bf16*)(ws + alloc((size_t)2 * DI * DM * 2));
  __bf16* xprojbf  = (__bf16*)(ws + alloc((size_t)NK * 64 * DI * 2));
  __bf16* outwbf   = (__bf16*)(ws + alloc((size_t)DM * DI * 2));
  float*  x1       = (float*)(ws + alloc((size_t)NB * L0 * DI * 4));
  float*  zbuf     = (float*)(ws + alloc((size_t)NB * L0 * DI * 4));
  float*  convf    = (float*)(ws + alloc((size_t)NB * L0 * DI * 4));
  __bf16* convbf   = (__bf16*)(ws + alloc((size_t)NB * L0 * DI * 2));
  float*  xdbl     = (float*)(ws + alloc((size_t)NB * NK * 64 * L0 * 4));
  float*  ybuf     = (float*)(ws + alloc((size_t)NB * L0 * DI * 4));
  __bf16* yg       = (__bf16*)(ws + alloc((size_t)NB * L0 * DI * 2));
  (void)ws_size; (void)in_sizes; (void)n_in; (void)out_size;

  // 1) conversions to bf16
  { int n = NB * L0 * DM;      cvt_bf16<<<(n + 255) / 256, 256, 0, stream>>>(x_f, x_bf, n); }
  { int n = 2 * DI * DM;       cvt_bf16<<<(n + 255) / 256, 256, 0, stream>>>(inw_f, winbf, n); }
  { int n = NK * 64 * DI;      cvt_bf16<<<(n + 255) / 256, 256, 0, stream>>>(xproj_f, xprojbf, n); }
  { int n = DM * DI;           cvt_bf16<<<(n + 255) / 256, 256, 0, stream>>>(outw_f, outwbf, n); }

  // 2) in_proj GEMM (WMMA, 16x64 strips), split x1/z
  { int strips = (NB * L0 / 16) * (2 * DI / 64);
    gemm_inproj<<<(strips + 7) / 8, 256, 0, stream>>>(x_bf, winbf, x1, zbuf); }

  // 3) depthwise conv + SiLU
  conv_silu<<<NB * L0, 256, 0, stream>>>(x1, convw_f, convb_f, convf, convbf);

  // 4) x_dbl GEMM (WMMA) with direction permutation folded into B addressing
  { int strips = 16 * 4 * 16;
    gemm_xdbl<<<(strips + 7) / 8, 256, 0, stream>>>(xprojbf, convbf, xdbl); }

  // 5) zero combined-y accumulator, then selective scan with fused combine
  { int n = NB * L0 * DI; zero_f32<<<(n + 255) / 256, 256, 0, stream>>>(ybuf, n); }
  scan_kernel<<<NB * NK * (DI / 256), 256, 0, stream>>>(xdbl, convf, dtw_f, dtb_f,
                                                        alogs_f, ds_f, ybuf);

  // 6) LayerNorm + SiLU gate -> bf16
  ln_gate<<<NB * L0, 256, 0, stream>>>(ybuf, zbuf, lng_f, lnb_f, yg);

  // 7) out_proj GEMM (WMMA, 16x64 strips) + bias
  { int strips = (NB * L0 / 16) * (DM / 64);
    gemm_out<<<(strips + 7) / 8, 256, 0, stream>>>(yg, outwbf, outb_f, out); }
}
